// folibiKT_63539746177588
// MI455X (gfx1250) — compile-verified, matrix operand order checked by reading, not run on hardware
//
#include <hip/hip_runtime.h>
#include <hip/hip_bf16.h>

#define BB 8
#define SS 1024
#define EE 512
#define HH 8
#define DD 64
#define PFP (SS + 8)   // padded f16 P row length (breaks LDS bank alignment)

typedef __attribute__((ext_vector_type(16))) _Float16 v16h;
typedef __attribute__((ext_vector_type(8)))  _Float16 v8h;
typedef __attribute__((ext_vector_type(8)))  float    v8f;

// ---------------------------------------------------------------------------
// WMMA 16x16x32 f16 fragment layout (ISA 7.12.2):
//   A (16x32): lanes 0-15 hold M=lane, K={0..7,16..23}; lanes 16-31 hold
//              M=lane-16, K={8..15,24..31}  (two contiguous 8-half runs)
//   B (32x16): symmetric with N in place of M -> lane holds row N of B^T,
//              so out = X * W^T loads W rows directly as the B fragment.
// ---------------------------------------------------------------------------
__device__ inline v8f wmma16(v16h a, v16h b, v8f c) {
  return __builtin_amdgcn_wmma_f32_16x16x32_f16(
      false, a, false, b, (short)0, c, false, false);
}

// fragment from an fp32 row (row-major, contracted over last dim)
__device__ inline v16h frag_f32(const float* row, int kbase, int hi) {
  const float4* p0 = reinterpret_cast<const float4*>(row + kbase + hi * 8);
  const float4* p1 = reinterpret_cast<const float4*>(row + kbase + 16 + hi * 8);
  float4 a0 = p0[0], a1 = p0[1], b0 = p1[0], b1 = p1[1];
  v16h f;
  f[0]  = (_Float16)a0.x; f[1]  = (_Float16)a0.y; f[2]  = (_Float16)a0.z; f[3]  = (_Float16)a0.w;
  f[4]  = (_Float16)a1.x; f[5]  = (_Float16)a1.y; f[6]  = (_Float16)a1.z; f[7]  = (_Float16)a1.w;
  f[8]  = (_Float16)b0.x; f[9]  = (_Float16)b0.y; f[10] = (_Float16)b0.z; f[11] = (_Float16)b0.w;
  f[12] = (_Float16)b1.x; f[13] = (_Float16)b1.y; f[14] = (_Float16)b1.z; f[15] = (_Float16)b1.w;
  return f;
}

// fragment from an f16 row (row-major, contracted over last dim)
__device__ inline v16h frag_f16(const _Float16* row, int kbase, int hi) {
  v8h lo = *reinterpret_cast<const v8h*>(row + kbase + hi * 8);
  v8h up = *reinterpret_cast<const v8h*>(row + kbase + 16 + hi * 8);
  v16h f;
#pragma unroll
  for (int t = 0; t < 8; ++t) { f[t] = lo[t]; f[8 + t] = up[t]; }
  return f;
}

// ---------------------------------------------------------------------------
// 1) Input projections. qh,kh use Wk/bk (kq_same), vh uses Wv/bv.
//    qh,kh,vh stored f16 [B,H,S,D]; V additionally stored transposed as
//    Vt [B,H,D,S] so the attention P@V B-fragments are contiguous loads.
// ---------------------------------------------------------------------------
__global__ __launch_bounds__(256) void proj_kernel(
    const float* __restrict__ q, const float* __restrict__ k,
    const float* __restrict__ v, const float* __restrict__ Wk,
    const float* __restrict__ bk, const float* __restrict__ Wv,
    const float* __restrict__ bv, _Float16* __restrict__ qh,
    _Float16* __restrict__ kh, _Float16* __restrict__ vh,
    _Float16* __restrict__ vt) {
  const int lane = threadIdx.x & 31;
  const int wv   = threadIdx.x >> 5;
  const int hi   = lane >> 4;
  const int ln   = lane & 15;
  const int z    = blockIdx.y;

  const float* X    = (z == 0) ? q : ((z == 1) ? k : v);
  const float* W    = (z == 2) ? Wv : Wk;
  const float* bias = (z == 2) ? bv : bk;
  _Float16*    Out  = (z == 0) ? qh : ((z == 1) ? kh : vh);

  const int tile = blockIdx.x * 8 + wv;   // 512 M-tiles x 32 N-tiles
  const int tM = tile >> 5;
  const int tN = tile & 31;

  const float* xrow = X + (size_t)(tM * 16 + ln) * EE;
  const float* wrow = W + (size_t)(tN * 16 + ln) * EE;

  v8f c = {};
#pragma unroll 4
  for (int kb = 0; kb < EE; kb += 32) {
    v16h a = frag_f32(xrow, kb, hi);
    v16h b = frag_f32(wrow, kb, hi);
    c = wmma16(a, b, c);
  }

  const int col = tN * 16 + ln;
  const float bia = bias[col];
  const int hh = col >> 6;   // head
  const int dd = col & 63;   // dim within head
#pragma unroll
  for (int vi = 0; vi < 8; ++vi) {
    const int mr   = tM * 16 + vi + hi * 8;  // flat (b*S+s)
    const int bidx = mr >> 10;
    const int srow = mr & (SS - 1);
    const _Float16 val = (_Float16)(c[vi] + bia);
    Out[(((size_t)bidx * HH + hh) * SS + srow) * DD + dd] = val;
    if (z == 2)  // transposed copy of V: [B,H,D,S]
      vt[(((size_t)bidx * HH + hh) * DD + dd) * SS + srow] = val;
  }
}

// ---------------------------------------------------------------------------
// 2) Attention with distance decay. One block per (b,h, 16-row block).
//    512 threads = 16 waves. LDS: scores fp32 64KB + P f16 ~32KB + acc 4KB.
// ---------------------------------------------------------------------------
__global__ __launch_bounds__(512) void attn_kernel(
    const _Float16* __restrict__ qh, const _Float16* __restrict__ kh,
    const _Float16* __restrict__ vt, const float* __restrict__ gammas,
    _Float16* __restrict__ attn) {
  __shared__ float    sc[16][SS];    // raw / decayed scores
  __shared__ _Float16 pf[16][PFP];   // final probabilities (f16, padded rows)
  __shared__ float    acc[16][DD];   // P@V reduction accumulator

  const int lane = threadIdx.x & 31;
  const int w    = threadIdx.x >> 5;
  const int hi   = lane >> 4;
  const int ln   = lane & 15;

  const int bh = blockIdx.y;          // 0..63
  const int b  = bh >> 3;
  const int h  = bh & 7;
  const int i0 = blockIdx.x * 16;

  const _Float16* Q  = qh + (size_t)bh * SS * DD;
  const _Float16* K  = kh + (size_t)bh * SS * DD;
  const _Float16* Vt = vt + (size_t)bh * DD * SS;

  const int numJT = (i0 >> 4) + 1;        // causal tile count
  const int numJB = (numJT + 1) >> 1;     // 32-wide K blocks for P@V
  const float scale = 0.125f;             // 1/sqrt(64)

  // ---- stage 1: raw scores -> LDS --------------------------------------
  const _Float16* qrow = Q + (size_t)(i0 + ln) * DD;
  for (int jt = w; jt < numJT; jt += 16) {
    const _Float16* krow = K + (size_t)(jt * 16 + ln) * DD;
    v8f c = {};
    c = wmma16(frag_f16(qrow, 0, hi),  frag_f16(krow, 0, hi),  c);
    c = wmma16(frag_f16(qrow, 32, hi), frag_f16(krow, 32, hi), c);
#pragma unroll
    for (int vi = 0; vi < 8; ++vi)
      sc[vi + hi * 8][jt * 16 + ln] = c[vi] * scale;
  }
  __syncthreads();

  // ---- stage 2: softmax-1 -> prefix scan (distcum) -> decay -> softmax-2
  {
    const int r = w;            // wave w owns row r
    const int i = i0 + r;       // global query index
    const float g = gammas[h];
    const float gamma = -(fmaxf(g, 0.f) + log1pf(__expf(-fabsf(g))));  // -softplus

    float m1 = -3.0e38f;
    for (int j = lane; j <= i; j += 32) m1 = fmaxf(m1, sc[r][j]);
    for (int off = 16; off; off >>= 1) m1 = fmaxf(m1, __shfl_xor(m1, off, 32));
    float Z1 = 0.f;
    for (int j = lane; j <= i; j += 32) Z1 += __expf(sc[r][j] - m1);
    for (int off = 16; off; off >>= 1) Z1 += __shfl_xor(Z1, off, 32);
    const float invZ1 = 1.0f / Z1;

    float carry = 0.f;
    float m2 = -3.0e38f;
    for (int jb = 0; jb * 32 <= i; ++jb) {
      const int j = jb * 32 + lane;
      const bool ok = (j <= i);
      const float sraw = ok ? sc[r][j] : 0.f;
      float p = ok ? __expf(sraw - m1) : 0.f;
#pragma unroll
      for (int off = 1; off < 32; off <<= 1) {   // wave inclusive scan
        float t = __shfl_up(p, off, 32);
        if (lane >= off) p += t;
      }
      const float prefix = carry + p;            // inclusive prefix of e
      carry += __shfl(p, 31, 32);                // chunk total
      if (ok) {
        const float rem  = 1.0f - prefix * invZ1;     // disttot - distcum
        const float posd = (float)(i - j);
        const float dist = sqrtf(fmaxf(rem * posd, 0.f));
        float te = __expf(gamma * dist);
        te = fminf(fmaxf(te, 1e-5f), 1e5f);
        const float s2 = sraw * te;
        sc[r][j] = s2;
        m2 = fmaxf(m2, s2);
      }
    }
    for (int off = 16; off; off >>= 1) m2 = fmaxf(m2, __shfl_xor(m2, off, 32));

    float Z2 = 0.f;
    for (int j = lane; j <= i; j += 32) Z2 += __expf(sc[r][j] - m2);
    for (int off = 16; off; off >>= 1) Z2 += __shfl_xor(Z2, off, 32);
    const float invZ2 = (i == 0) ? 0.f : (1.0f / Z2);  // zero_pad row 0
    const int jLim = numJB * 32;
    for (int j = lane; j < jLim; j += 32)
      pf[r][j] = (j <= i) ? (_Float16)(__expf(sc[r][j] - m2) * invZ2)
                          : (_Float16)0.f;
  }
  __syncthreads();

  // ---- stage 3: out = P @ V (P from LDS f16, V^T rows from Vt) ---------
  for (int t = threadIdx.x; t < 16 * DD; t += 512) (&acc[0][0])[t] = 0.f;
  __syncthreads();

  {
    const int dt = w & 3;                  // d-tile 0..3
    const int n  = dt * 16 + ln;           // output column within D
    const _Float16* vtrow = Vt + (size_t)n * SS;
    v8f c = {};
    for (int jb = (w >> 2); jb < numJB; jb += 4) {
      // A fragment: two contiguous 16B LDS loads from padded P rows
      v8h lo = *reinterpret_cast<const v8h*>(&pf[ln][jb * 32 + hi * 8]);
      v8h up = *reinterpret_cast<const v8h*>(&pf[ln][jb * 32 + 16 + hi * 8]);
      v16h a;
#pragma unroll
      for (int t = 0; t < 8; ++t) { a[t] = lo[t]; a[8 + t] = up[t]; }
      // B fragment: contiguous 16B global loads from transposed V
      v16h bf = frag_f16(vtrow, jb * 32, hi);
      c = wmma16(a, bf, c);
    }
#pragma unroll
    for (int vi = 0; vi < 8; ++vi)
      atomicAdd(&acc[vi + hi * 8][n], c[vi]);   // ds_add_f32 reduction
  }
  __syncthreads();

  // ---- stage 4: store merged-head f16 [B,S,E] --------------------------
  for (int t = threadIdx.x; t < 16 * DD; t += 512) {
    const int r  = t >> 6;
    const int dd = t & 63;
    attn[((size_t)(b * SS + i0 + r)) * EE + h * DD + dd] =
        (_Float16)acc[r][dd];
  }
}

// ---------------------------------------------------------------------------
// 3) Output projection: out = attn @ Wo^T + bo (fp32 result)
// ---------------------------------------------------------------------------
__global__ __launch_bounds__(256) void out_proj_kernel(
    const _Float16* __restrict__ attn, const float* __restrict__ Wo,
    const float* __restrict__ bo, float* __restrict__ out) {
  const int lane = threadIdx.x & 31;
  const int wv   = threadIdx.x >> 5;
  const int hi   = lane >> 4;
  const int ln   = lane & 15;

  const int tile = blockIdx.x * 8 + wv;
  const int tM = tile >> 5;
  const int tN = tile & 31;

  const _Float16* arow = attn + (size_t)(tM * 16 + ln) * EE;
  const float*    wrow = Wo + (size_t)(tN * 16 + ln) * EE;

  v8f c = {};
#pragma unroll 4
  for (int kb = 0; kb < EE; kb += 32) {
    v16h a = frag_f16(arow, kb, hi);
    v16h b = frag_f32(wrow, kb, hi);
    c = wmma16(a, b, c);
  }

  const int col = tN * 16 + ln;
  const float bia = bo[col];
#pragma unroll
  for (int vi = 0; vi < 8; ++vi) {
    const int mr = tM * 16 + vi + hi * 8;
    out[(size_t)mr * EE + col] = c[vi] + bia;
  }
}

// ---------------------------------------------------------------------------
extern "C" void kernel_launch(void* const* d_in, const int* in_sizes, int n_in,
                              void* d_out, int out_size, void* d_ws,
                              size_t ws_size, hipStream_t stream) {
  (void)in_sizes; (void)n_in; (void)out_size; (void)ws_size;
  const float* q      = (const float*)d_in[0];
  const float* k      = (const float*)d_in[1];
  const float* v      = (const float*)d_in[2];
  const float* Wk     = (const float*)d_in[3];
  const float* bk     = (const float*)d_in[4];
  const float* Wv     = (const float*)d_in[5];
  const float* bv     = (const float*)d_in[6];
  const float* Wo     = (const float*)d_in[7];
  const float* bo     = (const float*)d_in[8];
  const float* gammas = (const float*)d_in[9];

  const size_t elems = (size_t)BB * SS * EE;   // 4M f16 per buffer
  _Float16* qh   = (_Float16*)d_ws;
  _Float16* kh   = qh + elems;
  _Float16* vh   = kh + elems;
  _Float16* vt   = vh + elems;   // V transposed [B,H,D,S]
  _Float16* attn = vt + elems;

  proj_kernel<<<dim3(2048, 3), 256, 0, stream>>>(q, k, v, Wk, bk, Wv, bv,
                                                 qh, kh, vh, vt);
  attn_kernel<<<dim3(SS / 16, BB * HH), 512, 0, stream>>>(qh, kh, vt, gammas,
                                                          attn);
  out_proj_kernel<<<dim3(2048), 256, 0, stream>>>(attn, Wo, bo,
                                                  (float*)d_out);
}
